// Head_26542897889536
// MI455X (gfx1250) — matrix operand order repaired and checked by reading, not compiled, not run on hardware
//
#include <hip/hip_runtime.h>
#include <math.h>

// ---------------------------------------------------------------------------
// Single-head causal attention, CDNA5 (gfx1250) wave32 WMMA implementation.
//   B=4, T=4096, C=512, H=64.  out = softmax(mask(QK^T * C^-0.5)) V, fp32.
// Phase 1: QKV projection (bf16 WMMA, fp32 accum) -> ws (Q,K row-major, V^T).
// Phase 2: flash attention, one wave per 16-query tile, 32-key blocks.
// ---------------------------------------------------------------------------

typedef __attribute__((ext_vector_type(16))) __bf16 v16bf;
typedef __attribute__((ext_vector_type(8)))  float  v8f;

#define C_DIM 512
#define H_DIM 64

static __device__ __forceinline__ __bf16 f2bf(float f) { return (__bf16)f; }

// ---------------------------------------------------------------------------
// Phase 1: q/k/v = x @ W{q,k,v}^T.   x:[M,C] f32, W:[H,C] f32.
// Q,K -> bf16 [M,H] row-major.  V -> bf16 [B,H,T] (transposed) for phase 2.
// One wave handles a 16-row M-tile and all 3*4 = 12 N-tiles of 16 columns.
// ---------------------------------------------------------------------------
__global__ __launch_bounds__(256) void qkv_proj_kernel(
    const float* __restrict__ x,
    const float* __restrict__ Wq,
    const float* __restrict__ Wk,
    const float* __restrict__ Wv,
    __bf16* __restrict__ Qb,   // [M,H]
    __bf16* __restrict__ Kb,   // [M,H]
    __bf16* __restrict__ Vt,   // [B,H,T]
    int M, int T)
{
    const int lane = threadIdx.x & 31;
    const int wave = threadIdx.x >> 5;
    const int ln   = lane & 15;        // A-row / B-col / D-col index
    const int hi   = lane >> 4;        // half-wave select

    const int m0 = (blockIdx.x * 8 + wave) * 16;
    if (m0 >= M) return;               // wave-uniform

    v8f acc[12];
#pragma unroll
    for (int j = 0; j < 12; ++j)
#pragma unroll
        for (int r = 0; r < 8; ++r) acc[j][r] = 0.0f;

    const float* xrow = x + (size_t)(m0 + ln) * C_DIM;
    const float* Wall[3] = { Wq, Wk, Wv };

    for (int kc = 0; kc < C_DIM; kc += 32) {
        // A fragment: 16x32 tile of x, converted to bf16.
        v16bf a;
        {
            const int kb = kc + hi * 16;
#pragma unroll
            for (int e = 0; e < 8; ++e) {
                a[e]     = f2bf(xrow[kb + e]);
                a[e + 8] = f2bf(xrow[kb + 8 + e]);
            }
        }
#pragma unroll
        for (int w = 0; w < 3; ++w) {
#pragma unroll
            for (int nt = 0; nt < 4; ++nt) {
                // B fragment: B[k][n] = W[nt*16+n][kc+k]; lane: n=ln, k=e+16*hi
                const float* wr = Wall[w] + (size_t)(nt * 16 + ln) * C_DIM + kc + hi * 16;
                v16bf bf;
#pragma unroll
                for (int e = 0; e < 16; ++e) bf[e] = f2bf(wr[e]);
                const int j = w * 4 + nt;
                acc[j] = __builtin_amdgcn_wmma_f32_16x16x32_bf16(
                    false, a, false, bf, (short)0, acc[j], false, false);
            }
        }
    }

    // Store Q, K row-major bf16.
#pragma unroll
    for (int w = 0; w < 2; ++w) {
        __bf16* dst = (w == 0) ? Qb : Kb;
#pragma unroll
        for (int nt = 0; nt < 4; ++nt) {
            const int col = nt * 16 + ln;
#pragma unroll
            for (int r = 0; r < 8; ++r) {
                const int m = m0 + r + 8 * hi;
                dst[(size_t)m * H_DIM + col] = f2bf(acc[w * 4 + nt][r]);
            }
        }
    }
    // Store V transposed: Vt[b][h][t], b = m/T, t = m%T (tile never crosses b).
#pragma unroll
    for (int nt = 0; nt < 4; ++nt) {
        const int col = nt * 16 + ln;
#pragma unroll
        for (int r = 0; r < 8; ++r) {
            const int m = m0 + r + 8 * hi;
            const int b = m / T, t = m % T;
            Vt[((size_t)b * H_DIM + col) * T + t] = f2bf(acc[8 + nt][r]);
        }
    }
}

// ---------------------------------------------------------------------------
// Phase 2: flash attention. One wave owns a 16-query tile; iterates 32-key
// blocks up to the causal frontier. Online softmax in fp32.
// ---------------------------------------------------------------------------
__global__ __launch_bounds__(128) void attn_kernel(
    const __bf16* __restrict__ Qb,   // [B,T,H]
    const __bf16* __restrict__ Kb,   // [B,T,H]
    const __bf16* __restrict__ Vt,   // [B,H,T]
    float* __restrict__ out,         // [B,T,H]
    int T, float scale)
{
    __shared__ __bf16 lds_p[4][16 * 32];   // per-wave P repack scratch (1KB ea)

    const int lane = threadIdx.x & 31;
    const int wave = threadIdx.x >> 5;
    const int ln   = lane & 15;
    const int hi   = lane >> 4;

    const int tilesPerBatch = T / 16;
    const int tileIdx = blockIdx.x * 4 + wave;
    const int b  = tileIdx / tilesPerBatch;
    const int q0 = (tileIdx % tilesPerBatch) * 16;

    const __bf16* Qbase = Qb + (size_t)b * T * H_DIM;
    const __bf16* Kbase = Kb + (size_t)b * T * H_DIM;
    const __bf16* Vbase = Vt + (size_t)b * H_DIM * T;

    // Q A-fragments for the whole tile: 16x64 = two K-chunks of 32.
    v16bf qa[2];
    {
        const __bf16* qrow = Qbase + (size_t)(q0 + ln) * H_DIM;
#pragma unroll
        for (int c = 0; c < 2; ++c) {
            const int kb = c * 32 + hi * 16;
#pragma unroll
            for (int e = 0; e < 8; ++e) {
                qa[c][e]     = qrow[kb + e];
                qa[c][e + 8] = qrow[kb + 8 + e];
            }
        }
    }

    float m_run[8], l_run[8];
    v8f o_acc[4];
#pragma unroll
    for (int r = 0; r < 8; ++r) { m_run[r] = -1e30f; l_run[r] = 0.0f; }
#pragma unroll
    for (int nt = 0; nt < 4; ++nt)
#pragma unroll
        for (int r = 0; r < 8; ++r) o_acc[nt][r] = 0.0f;

    const int kend = q0 + 16;                 // causal key bound (exclusive)
    for (int j0 = 0; j0 < kend; j0 += 32) {
        // ---- S = Q K^T for 32 keys (two 16-key N-tiles) ----
        v8f s[2];
#pragma unroll
        for (int t = 0; t < 2; ++t)
#pragma unroll
            for (int r = 0; r < 8; ++r) s[t][r] = 0.0f;
#pragma unroll
        for (int t = 0; t < 2; ++t) {
            const int kb0 = j0 + t * 16;
#pragma unroll
            for (int c = 0; c < 2; ++c) {
                // B[k][n] = K[kb0+n][c*32+k]; lane: n=ln, k=e+16*hi -> contiguous
                const __bf16* krow = Kbase + (size_t)(kb0 + ln) * H_DIM + c * 32 + hi * 16;
                v16bf kf;
#pragma unroll
                for (int e = 0; e < 16; ++e) kf[e] = krow[e];
                s[t] = __builtin_amdgcn_wmma_f32_16x16x32_bf16(
                    false, qa[c], false, kf, (short)0, s[t], false, false);
            }
        }

        // ---- scale + causal mask ----
#pragma unroll
        for (int t = 0; t < 2; ++t) {
            const int kp = j0 + t * 16 + ln;
#pragma unroll
            for (int r = 0; r < 8; ++r) {
                const int qp = q0 + r + 8 * hi;
                const float v = s[t][r] * scale;
                s[t][r] = (kp <= qp) ? v : -1e30f;
            }
        }

        // ---- online softmax: row max / exp / row sum (rows live in 16-lane half)
        float alpha[8];
#pragma unroll
        for (int r = 0; r < 8; ++r) {
            float mx = fmaxf(s[0][r], s[1][r]);
#pragma unroll
            for (int d = 1; d < 16; d <<= 1) mx = fmaxf(mx, __shfl_xor(mx, d, 32));
            const float mn = fmaxf(m_run[r], mx);
            alpha[r] = __expf(m_run[r] - mn);
            m_run[r] = mn;
            const float p0 = __expf(s[0][r] - mn);
            const float p1 = __expf(s[1][r] - mn);
            s[0][r] = p0; s[1][r] = p1;
            float sm = p0 + p1;
#pragma unroll
            for (int d = 1; d < 16; d <<= 1) sm += __shfl_xor(sm, d, 32);
            l_run[r] = l_run[r] * alpha[r] + sm;
        }

        // ---- rescale O, repack P (C/D layout -> A layout) via per-wave LDS ----
#pragma unroll
        for (int nt = 0; nt < 4; ++nt)
#pragma unroll
            for (int r = 0; r < 8; ++r) o_acc[nt][r] *= alpha[r];

        __bf16* pl = &lds_p[wave][0];
#pragma unroll
        for (int t = 0; t < 2; ++t)
#pragma unroll
            for (int r = 0; r < 8; ++r)
                pl[(r + 8 * hi) * 32 + t * 16 + ln] = f2bf(s[t][r]);
        // same-wave DS ordering; make stores visible and fence the compiler
        asm volatile("s_wait_dscnt 0" ::: "memory");

        v16bf pa;
        {
            const __bf16* prow = pl + ln * 32 + hi * 16;
#pragma unroll
            for (int e = 0; e < 8; ++e) {
                pa[e]     = prow[e];
                pa[e + 8] = prow[8 + e];
            }
        }
        asm volatile("s_wait_dscnt 0" ::: "memory");

        // ---- O += P V : B[k][n] = V[j0+k][nt*16+n] from V^T (contiguous) ----
#pragma unroll
        for (int nt = 0; nt < 4; ++nt) {
            const __bf16* vrow = Vbase + (size_t)(nt * 16 + ln) * T + j0 + 16 * hi;
            v16bf vf;
#pragma unroll
            for (int e = 0; e < 16; ++e) vf[e] = vrow[e];
            o_acc[nt] = __builtin_amdgcn_wmma_f32_16x16x32_bf16(
                false, pa, false, vf, (short)0, o_acc[nt], false, false);
        }
    }

    // ---- normalize and store fp32 output ----
    float inv_l[8];
#pragma unroll
    for (int r = 0; r < 8; ++r) inv_l[r] = 1.0f / l_run[r];

    float* orow = out + ((size_t)b * T + q0) * H_DIM;
#pragma unroll
    for (int nt = 0; nt < 4; ++nt) {
        const int col = nt * 16 + ln;
#pragma unroll
        for (int r = 0; r < 8; ++r)
            orow[(size_t)(r + 8 * hi) * H_DIM + col] = o_acc[nt][r] * inv_l[r];
    }
}

// ---------------------------------------------------------------------------
extern "C" void kernel_launch(void* const* d_in, const int* in_sizes, int n_in,
                              void* d_out, int out_size, void* d_ws, size_t ws_size,
                              hipStream_t stream) {
    const float* x  = (const float*)d_in[0];
    const float* Wq = (const float*)d_in[1];
    const float* Wk = (const float*)d_in[2];
    const float* Wv = (const float*)d_in[3];
    float* out = (float*)d_out;

    const int B = 4, T = 4096, H = H_DIM;
    const int M = B * T;                     // 16384 rows
    const float scale = 1.0f / sqrtf((float)C_DIM);

    // Workspace: Q[M,H] + K[M,H] + V^T[B,H,T], all bf16 (2 MB each = 6 MB).
    __bf16* Qb = (__bf16*)d_ws;
    __bf16* Kb = Qb + (size_t)M * H;
    __bf16* Vt = Kb + (size_t)M * H;

    // Phase 1: 1024 wave-tiles of 16 rows, 8 waves (256 thr) per block.
    qkv_proj_kernel<<<dim3((M / 16) / 8), dim3(256), 0, stream>>>(
        x, Wq, Wk, Wv, Qb, Kb, Vt, M, T);

    // Phase 2: 1024 query tiles, 4 waves (128 thr) per block.
    attn_kernel<<<dim3((M / 16) / 4), dim3(128), 0, stream>>>(
        Qb, Kb, Vt, out, T, scale);
}
